// PPIPredictor_68564857913750
// MI455X (gfx1250) — matrix operand order repaired and checked by reading, not compiled, not run on hardware
//
#include <hip/hip_runtime.h>
#include <math.h>
#include <stdint.h>

typedef __attribute__((ext_vector_type(16))) __bf16 v16bf;
typedef __attribute__((ext_vector_type(8)))  float  v8f;
typedef __attribute__((ext_vector_type(4)))  unsigned int u32x4;
typedef __attribute__((ext_vector_type(8)))  int i32x8;
typedef __attribute__((ext_vector_type(4)))  int i32x4;

#define HDIM 512
#define SDIM 512
#define NBATCH 2
#define NDIM 256   // H/2

// ---------------------------------------------------------------------------
// Prep 1: a'[m,k] = b1[k] + sum_h f[m,h]*W1[h,k]   (m = b*S+s, 1024 rows)
//         c [m,k] =         sum_h f[m,h]*W1[H+h,k]
// ---------------------------------------------------------------------------
__global__ __launch_bounds__(256) void prep_ac(
    const float* __restrict__ feats, const float* __restrict__ W1,
    const float* __restrict__ b1,
    float* __restrict__ A, float* __restrict__ C)
{
    __shared__ float fr[HDIM];
    const int m = blockIdx.x;
    const int t = threadIdx.x;
    fr[t]       = feats[m * HDIM + t];
    fr[t + 256] = feats[m * HDIM + t + 256];
    __syncthreads();

    float a0 = 0.f, a1 = 0.f, c0 = 0.f, c1 = 0.f;
    for (int h = 0; h < HDIM; ++h) {
        const float f  = fr[h];
        const float* wA = W1 + (size_t)h * HDIM;
        const float* wC = W1 + (size_t)(HDIM + h) * HDIM;
        a0 = fmaf(f, wA[t],       a0);
        a1 = fmaf(f, wA[t + 256], a1);
        c0 = fmaf(f, wC[t],       c0);
        c1 = fmaf(f, wC[t + 256], c1);
    }
    A[(size_t)m * HDIM + t]       = a0 + b1[t];
    A[(size_t)m * HDIM + t + 256] = a1 + b1[t + 256];
    C[(size_t)m * HDIM + t]       = c0;
    C[(size_t)m * HDIM + t + 256] = c1;
}

// ---------------------------------------------------------------------------
// Prep 2: W2t[n][k] = bf16(W2[k][n])  (B-fragment-friendly: 16 contiguous
// K-values = one 32B load per lane per WMMA).
// ---------------------------------------------------------------------------
__global__ __launch_bounds__(256) void prep_w2t(
    const float* __restrict__ W2, __bf16* __restrict__ W2t)
{
    const int k = blockIdx.x;    // 0..511
    const int n = threadIdx.x;   // 0..255
    W2t[(size_t)n * HDIM + k] = (__bf16)W2[(size_t)k * NDIM + n];
}

// ---------------------------------------------------------------------------
// TDM: DMA one K-slice of W2t (tile = 256 rows x 32 cols bf16 = 16 KB) into
// LDS at ldsaddr. D# per CDNA5 ISA §8.3/8.4:
//   tensor = [dim1=256][dim0=512] bf16 (data_size=2B), row stride 512
//   tile   = rows 0..255, cols kc*32..kc*32+31 -> packed 64B/row in LDS.
// ---------------------------------------------------------------------------
__device__ __forceinline__ void tdm_load_w2_slice(
    const __bf16* __restrict__ W2t, unsigned ldsaddr, int kc)
{
    const unsigned long long ga =
        (unsigned long long)(uintptr_t)W2t + (unsigned long long)kc * 64ull;
    u32x4 g0;
    g0[0] = 1u;                                   // count=1 valid user desc
    g0[1] = ldsaddr;                              // lds_addr (bytes)
    g0[2] = (unsigned)ga;                         // global_addr[31:0]
    g0[3] = (unsigned)((ga >> 32) & 0x01FFFFFFu)  // global_addr[56:32]
          | 0x80000000u;                          // type=2 ("image")
    i32x8 g1;
    g1[0] = 0x00010000;          // workgroup_mask=0, data_size=1 (2 bytes)
    g1[1] = (int)(512u << 16);   // tensor_dim0[15:0]=512  (bits 63:48)
    g1[2] = (int)(256u << 16);   // tensor_dim1[15:0]=256  (bits 95:80)
    g1[3] = (int)(32u  << 16);   // tile_dim0=32           (bits 127:112)
    g1[4] = 256;                 // tile_dim1=256, tile_dim2=0
    g1[5] = 512;                 // tensor_dim0_stride[31:0]=512
    g1[6] = 0;                   // stride msbs / dim1_stride lo
    g1[7] = 0;
    const i32x4 gz = {0, 0, 0, 0};
#if __clang_major__ >= 23
    const i32x8 gz8 = {0, 0, 0, 0, 0, 0, 0, 0};
    __builtin_amdgcn_tensor_load_to_lds(g0, g1, gz, gz, gz8, 0);
#else
    __builtin_amdgcn_tensor_load_to_lds(g0, g1, gz, gz, 0);
#endif
}

// ---------------------------------------------------------------------------
// Main fused kernel. WG = 256 threads (8 waves) = one (b,i), 128 j's.
// Wave w: 16-pair M-tile. D = relu(a'_i + c_j) @ W2 via
// v_wmma_f32_16x16x32_bf16 with TDM-double-buffered W2 slices in LDS.
// ---------------------------------------------------------------------------
__global__ __launch_bounds__(256) void pair_mlp(
    const float* __restrict__ A, const float* __restrict__ C,
    const __bf16* __restrict__ W2t,
    const float* __restrict__ b2, const float* __restrict__ W3,
    const float* __restrict__ b3, float* __restrict__ out)
{
    __shared__ __align__(128) __bf16 w2s[2][NDIM * 32];  // 2 x 16 KB
    __shared__ __align__(16)  float  ar[HDIM];           // a' row for (b,i)

    const int bx   = blockIdx.x;
    const int jblk = bx & 3;
    const int i    = (bx >> 2) & (SDIM - 1);
    const int b    = bx >> 11;
    const int tid  = threadIdx.x;
    const int wave = tid >> 5;
    const int lane = tid & 31;
    const int col  = lane & 15;   // A row m / D column / C-row owner
    const int kgrp = lane >> 4;   // lane half: selects K sub-runs

    // stage a' row (shared by all 8 waves)
    {
        const float* ap = A + ((size_t)b * SDIM + i) * HDIM;
        ar[tid]       = ap[tid];
        ar[tid + 256] = ap[tid + 256];
    }

    const int j0 = jblk * 128 + wave * 16;
    const float* cP = C + ((size_t)b * SDIM + j0 + col) * HDIM;

    const unsigned lds0 = (unsigned)(uintptr_t)&w2s[0][0];
    const unsigned lds1 = (unsigned)(uintptr_t)&w2s[1][0];

    // Kick off TDM for the first two K-slices (wave 0 only; EXEC ignored).
    if (tid < 32) {
        tdm_load_w2_slice(W2t, lds0, 0);
        tdm_load_w2_slice(W2t, lds1, 1);
    }

    v8f acc[16];
#pragma unroll
    for (int t = 0; t < 16; ++t) acc[t] = (v8f){0.f,0.f,0.f,0.f,0.f,0.f,0.f,0.f};

    for (int kc = 0; kc < 16; ++kc) {
        // Issuing wave drains TENSORcnt down to the still-in-flight prefetch,
        // then the barrier releases everyone onto the ready buffer.
        if (tid < 32) {
            if (kc < 15) __builtin_amdgcn_s_wait_tensorcnt(1);
            else         __builtin_amdgcn_s_wait_tensorcnt(0);
        }
        __syncthreads();

        const __bf16* wbuf = w2s[kc & 1];

        // A fragment: contiguous k order (B loads use the same).
        // half 0: K = kc*32 + {0..15}; half 1: +16.
        const int kb0 = kc * 32 + kgrp * 16;
        const int kb1 = kb0 + 8;
        float av[16], cv[16];
        *(float4*)(av + 0)  = *(const float4*)(ar + kb0);
        *(float4*)(av + 4)  = *(const float4*)(ar + kb0 + 4);
        *(float4*)(av + 8)  = *(const float4*)(ar + kb1);
        *(float4*)(av + 12) = *(const float4*)(ar + kb1 + 4);
        *(float4*)(cv + 0)  = *(const float4*)(cP + kb0);
        *(float4*)(cv + 4)  = *(const float4*)(cP + kb0 + 4);
        *(float4*)(cv + 8)  = *(const float4*)(cP + kb1);
        *(float4*)(cv + 12) = *(const float4*)(cP + kb1 + 4);
        v16bf af;
#pragma unroll
        for (int e = 0; e < 16; ++e)
            af[e] = (__bf16)fmaxf(av[e] + cv[e], 0.0f);

        // 16 N-tiles; same A fragment across all 16 WMMAs -> peel t=0
        // (reuse_a must be a frontend ICE), then reuse_a=1 for t=1..15.
        {
            const v16bf bf = *(const v16bf*)(wbuf + col * 32 + kgrp * 16);
            acc[0] = __builtin_amdgcn_wmma_f32_16x16x32_bf16(
                false, af, false, bf, (short)0, acc[0], false, false);
        }
#pragma unroll
        for (int t = 1; t < 16; ++t) {
            const v16bf bf = *(const v16bf*)(wbuf + (t * 16 + col) * 32 + kgrp * 16);
            acc[t] = __builtin_amdgcn_wmma_f32_16x16x32_bf16(
                false, af, false, bf, (short)0, acc[t], true, false);
        }

        __syncthreads();   // all waves done reading w2s[kc&1]
        if (tid < 32 && kc + 2 < 16)
            tdm_load_w2_slice(W2t, (kc & 1) ? lds1 : lds0, kc + 2);
    }

    // Epilogue: lane holds D rows m = 8*kgrp + v at column n = t*16 + col.
    const float b3v = b3[0];
    float sum[8] = {0.f,0.f,0.f,0.f,0.f,0.f,0.f,0.f};
#pragma unroll
    for (int t = 0; t < 16; ++t) {
        const int n = t * 16 + col;
        const float b2v = b2[n];
        const float w3v = W3[n];
#pragma unroll
        for (int v = 0; v < 8; ++v) {
            float h2 = fmaxf(acc[t][v] + b2v, 0.0f);
            sum[v] = fmaf(h2, w3v, sum[v]);
        }
    }
#pragma unroll
    for (int off = 1; off < 16; off <<= 1) {
#pragma unroll
        for (int v = 0; v < 8; ++v)
            sum[v] += __shfl_xor(sum[v], off, 32);
    }
    if (col == 0) {
        float* op = out + ((size_t)b * SDIM + i) * SDIM + j0 + kgrp * 8;
#pragma unroll
        for (int v = 0; v < 8; ++v)
            op[v] = 1.0f / (1.0f + __expf(-(sum[v] + b3v)));
    }
}

// ---------------------------------------------------------------------------
extern "C" void kernel_launch(void* const* d_in, const int* in_sizes, int n_in,
                              void* d_out, int out_size, void* d_ws, size_t ws_size,
                              hipStream_t stream)
{
    const float* feats = (const float*)d_in[0];
    const float* W1    = (const float*)d_in[1];
    const float* b1    = (const float*)d_in[2];
    const float* W2    = (const float*)d_in[3];
    const float* b2    = (const float*)d_in[4];
    const float* W3    = (const float*)d_in[5];
    const float* b3    = (const float*)d_in[6];
    float* out = (float*)d_out;

    char* ws = (char*)d_ws;
    const size_t acBytes = (size_t)NBATCH * SDIM * HDIM * sizeof(float); // 2 MB
    float*  Abuf = (float*)ws;
    float*  Cbuf = (float*)(ws + acBytes);
    __bf16* W2t  = (__bf16*)(ws + 2 * acBytes);                          // 256 KB

    prep_ac <<<NBATCH * SDIM, 256, 0, stream>>>(feats, W1, b1, Abuf, Cbuf);
    prep_w2t<<<HDIM,          256, 0, stream>>>(W2, W2t);
    pair_mlp<<<NBATCH * SDIM * (SDIM / 128), 256, 0, stream>>>(
        Abuf, Cbuf, W2t, b2, W3, b3, out);
}
